// WaveglowBlock_62895501083229
// MI455X (gfx1250) — matrix operand order repaired and checked by reading, not compiled, not run on hardware
//
#include <hip/hip_runtime.h>
#include <hip/hip_bf16.h>
#include <math.h>

typedef _Float16 half_t;
typedef __attribute__((ext_vector_type(16))) _Float16 v16h;
typedef __attribute__((ext_vector_type(8)))  _Float16 v8h;
typedef __attribute__((ext_vector_type(8)))  float    v8f;

#define NLAYERS 8
#define CH   256
#define MEL  640
#define BB   16
#define TT   8192
#define RTOT (BB*TT)          // 131072 rows
#define KA   (3*CH + MEL)     // 1408: [x(t-d) | x(t) | x(t+d) | spect(t)]
#define NOUT (2*CH)           // 512
#define KSTEPS (KA/32)        // 44 (even)

__device__ __forceinline__ v8h ld8(const half_t* p) { return *(const v8h*)p; }
__device__ __forceinline__ v16h mk16(v8h a, v8h b) {
  return __builtin_shufflevector(a, b, 0,1,2,3,4,5,6,7,8,9,10,11,12,13,14,15);
}

// ---------------- prep kernels ----------------

__global__ void wg_cvt_spect(const float* __restrict__ s, half_t* __restrict__ sh, long long n) {
  long long i = (long long)blockIdx.x * blockDim.x + threadIdx.x;
  if (i < n) sh[i] = (half_t)s[i];
}

// Wt_in[layer][cout(512)][K(1408)] ; K = tap0 cin 0..255, tap1, tap2, cond cin 0..639
__global__ void wg_prep_wtin(const float* __restrict__ in_w, const float* __restrict__ cond_w,
                             half_t* __restrict__ wt) {
  long long idx = (long long)blockIdx.x * blockDim.x + threadIdx.x;
  const long long total = (long long)NLAYERS * NOUT * KA;
  if (idx >= total) return;
  int i   = (int)(idx / ((long long)NOUT * KA));
  int rem = (int)(idx % ((long long)NOUT * KA));
  int co  = rem / KA;
  int k   = rem % KA;
  float v;
  if (k < 3*CH) {
    int tap = k / CH, cin = k % CH;
    v = in_w[(((size_t)i*3 + tap)*CH + cin)*NOUT + co];       // (8,3,256,512)
  } else {
    int cin = k - 3*CH;
    v = cond_w[((size_t)i*MEL + cin)*NOUT + co];              // (8,1,640,512)
  }
  wt[idx] = (half_t)v;
}

// Wt_rs[layer][cout(512)][cin(256)] ; layer 7 uses rs_last (only couts 0..255)
__global__ void wg_prep_wtrs(const float* __restrict__ rs_w, const float* __restrict__ rs_last_w,
                             half_t* __restrict__ wt) {
  long long idx = (long long)blockIdx.x * blockDim.x + threadIdx.x;
  const long long total = (long long)NLAYERS * NOUT * CH;
  if (idx >= total) return;
  int i   = (int)(idx / (NOUT*CH));
  int rem = (int)(idx % (NOUT*CH));
  int co  = rem / CH;
  int cin = rem % CH;
  float v = 0.0f;
  if (i < NLAYERS-1)      v = rs_w[((size_t)i*CH + cin)*NOUT + co];   // (7,1,256,512)
  else if (co < CH)       v = rs_last_w[(size_t)cin*CH + co];         // (1,256,256)
  wt[idx] = (half_t)v;
}

// start 1x1 conv: x[r][c] = sum_j audio[r][j]*start_w[j][c] + start_b[c]  (K=4, VALU)
__global__ void wg_start(const float* __restrict__ audio, const float* __restrict__ sw,
                         const float* __restrict__ sb, half_t* __restrict__ xh) {
  long long idx = (long long)blockIdx.x * blockDim.x + threadIdx.x;
  if (idx >= (long long)RTOT * CH) return;
  int r = (int)(idx >> 8);
  int c = (int)(idx & 255);
  float a = sb[c];
  #pragma unroll
  for (int j = 0; j < 4; ++j) a += audio[(size_t)r*4 + j] * sw[j*CH + c];
  xh[idx] = (half_t)a;
}

// ---------------- fused WaveNet layer (WMMA) ----------------
// block = 256 threads (8 waves), tile = 32 rows x full 512 cols.
// Each wave owns a unique 64-col strip across BOTH 16-row tiles.
// Stage A K-loop is software-pipelined with explicit double buffers so
// next-step A/B fragment loads overlap the current step's 8 WMMAs.
__global__ __launch_bounds__(256, 1)
void wg_layer(const half_t* __restrict__ x_in,
              half_t*       __restrict__ x_out,
              const half_t* __restrict__ spect_h,
              const half_t* __restrict__ WtIn,   // [512][1408]
              const half_t* __restrict__ WtRs,   // [512][256]
              const float*  __restrict__ in_b,   // [512]
              const float*  __restrict__ cond_b, // [512]
              const float*  __restrict__ rs_b,   // [512] (or [256] last)
              float*        __restrict__ skip,
              int dil, int is_first, int is_last)
{
  __shared__ __align__(16) half_t ia_lds[32*512];    // 32 KB
  __shared__ __align__(16) half_t acts_lds[32*256];  // 16 KB

  const int tid    = threadIdx.x;
  const int lane   = tid & 31;
  const int wv     = tid >> 5;
  const int lane16 = lane & 15;
  const int hi     = lane >> 4;

  const int row_base = blockIdx.x * 32;
  const int b      = row_base / TT;
  const int t_base = row_base - b*TT;

  const int nb  = wv * 64;              // stage-A col strip (4 tiles of 16)
  const int klo = hi * 8;               // A K sub-offset (ISA lane layout)
  const int tA0 = t_base + lane16;      // A row, row-tile 0
  const int tA1 = t_base + 16 + lane16; // A row, row-tile 1

  // fragment loaders for K-step ks
  auto loadA = [&](int ks, v16h afr[2]) {
    #pragma unroll
    for (int rt = 0; rt < 2; ++rt) {
      int tl = rt ? tA1 : tA0;
      v8h g0 = {}, g1 = {};
      if (ks < 24) {                                 // dilated x taps
        int tap  = ks >> 3;
        int cin  = (ks & 7) * 32;
        int trow = tl + (tap - 1) * dil;
        if (trow >= 0 && trow < TT) {                // SAME padding -> zeros
          const half_t* p = x_in + ((size_t)(b*TT + trow))*CH + cin + klo;
          g0 = ld8(p);
          g1 = ld8(p + 16);
        }
      } else {                                       // cond (spect) segment
        int off = (ks - 24) * 32;
        const half_t* p = spect_h + ((size_t)(b*TT + tl))*MEL + off + klo;
        g0 = ld8(p);
        g1 = ld8(p + 16);
      }
      afr[rt] = mk16(g0, g1);
    }
  };
  auto loadB = [&](int ks, v16h bfr[4]) {
    const half_t* wk = WtIn + (size_t)ks*32 + (size_t)hi*8;
    #pragma unroll
    for (int ct = 0; ct < 4; ++ct) {
      int co = nb + ct*16 + lane16;
      const half_t* wp = wk + (size_t)co * KA;
      bfr[ct] = mk16(ld8(wp), ld8(wp + 16));
    }
  };

  // ---- Stage A: in_act = A(32x1408) @ WtIn^T -> 32x512 ----
  v8f acc[2][4] = {};
  v16h a0[2], b0[4], a1[2], b1[4];
  loadA(0, a0);
  loadB(0, b0);
  #pragma unroll 1
  for (int ks = 0; ks < KSTEPS; ks += 2) {
    // prefetch ks+1 while computing ks
    loadA(ks + 1, a1);
    loadB(ks + 1, b1);
    #pragma unroll
    for (int ct = 0; ct < 4; ++ct) {
      acc[0][ct] = __builtin_amdgcn_wmma_f32_16x16x32_f16(
          false, a0[0], false, b0[ct], (short)0, acc[0][ct], false, false);
      acc[1][ct] = __builtin_amdgcn_wmma_f32_16x16x32_f16(
          false, a0[1], false, b0[ct], (short)0, acc[1][ct], false, false);
    }
    // prefetch ks+2 while computing ks+1 (tail reloads last step harmlessly)
    int k2 = (ks + 2 < KSTEPS) ? (ks + 2) : (KSTEPS - 1);
    loadA(k2, a0);
    loadB(k2, b0);
    #pragma unroll
    for (int ct = 0; ct < 4; ++ct) {
      acc[0][ct] = __builtin_amdgcn_wmma_f32_16x16x32_f16(
          false, a1[0], false, b1[ct], (short)0, acc[0][ct], false, false);
      acc[1][ct] = __builtin_amdgcn_wmma_f32_16x16x32_f16(
          false, a1[1], false, b1[ct], (short)0, acc[1][ct], false, false);
    }
  }

  // bias + stash in_act (f16) in LDS
  #pragma unroll
  for (int ct = 0; ct < 4; ++ct) {
    int n = nb + ct*16 + lane16;
    float bias = in_b[n] + cond_b[n];
    #pragma unroll
    for (int rt = 0; rt < 2; ++rt) {
      #pragma unroll
      for (int j = 0; j < 8; ++j) {
        int m = rt*16 + hi*8 + j;                    // C/D layout: VGPR j -> M=j(+8)
        ia_lds[m*512 + n] = (half_t)(acc[rt][ct][j] + bias);
      }
    }
  }
  __syncthreads();

  // gated activation: acts = tanh(ia[:,:256]) * sigmoid(ia[:,256:])
  for (int e = tid; e < 32*256; e += 256) {
    int m = e >> 8, c = e & 255;
    float gt = (float)ia_lds[m*512 + c];
    float gs = (float)ia_lds[m*512 + 256 + c];
    acts_lds[m*256 + c] = (half_t)(tanhf(gt) * (1.0f / (1.0f + __expf(-gs))));
  }
  __syncthreads();

  // ---- Stage B: rs = acts(32x256) @ WtRs^T -> 32 x nout2 ----
  const int nout2 = is_last ? CH : NOUT;
  const int nct2  = nout2 / 128;                     // col tiles per wave (4 or 2)
  const int nb2   = wv * (nout2 / 8);

  v8f acc2[2][4] = {};
  #pragma unroll 1
  for (int ks = 0; ks < CH/32; ++ks) {
    v16h afr[2];
    #pragma unroll
    for (int rt = 0; rt < 2; ++rt) {
      const half_t* ap = &acts_lds[(rt*16 + lane16)*256 + ks*32 + klo];
      afr[rt] = mk16(ld8(ap), ld8(ap + 16));
    }
    v16h bfr[4];
    for (int ct = 0; ct < nct2; ++ct) {
      int co = nb2 + ct*16 + lane16;
      const half_t* wp = WtRs + (size_t)co*CH + ks*32 + hi*8;
      bfr[ct] = mk16(ld8(wp), ld8(wp + 16));
    }
    for (int ct = 0; ct < nct2; ++ct) {
      acc2[0][ct] = __builtin_amdgcn_wmma_f32_16x16x32_f16(
          false, afr[0], false, bfr[ct], (short)0, acc2[0][ct], false, false);
      acc2[1][ct] = __builtin_amdgcn_wmma_f32_16x16x32_f16(
          false, afr[1], false, bfr[ct], (short)0, acc2[1][ct], false, false);
    }
  }

  // residual x update + skip accumulation
  for (int ct = 0; ct < nct2; ++ct) {
    int n = nb2 + ct*16 + lane16;
    float bias = rs_b[n];
    #pragma unroll
    for (int rt = 0; rt < 2; ++rt) {
      #pragma unroll
      for (int j = 0; j < 8; ++j) {
        int m = rt*16 + hi*8 + j;
        size_t r = (size_t)(row_base + m);
        float v = acc2[rt][ct][j] + bias;
        if (!is_last && n < CH) {
          x_out[r*CH + n] = (half_t)(v + (float)x_in[r*CH + n]);
        } else {
          int s = is_last ? n : (n - CH);
          if (is_first) skip[r*CH + s]  = v;
          else          skip[r*CH + s] += v;
        }
      }
    }
  }
}

// ---------------- end 1x1 conv: 256 -> 8 ----------------
__global__ void wg_end(const float* __restrict__ skip, const float* __restrict__ ew,
                       const float* __restrict__ eb, float* __restrict__ out) {
  int r = blockIdx.x * blockDim.x + threadIdx.x;
  if (r >= RTOT) return;
  float acc[8];
  #pragma unroll
  for (int o = 0; o < 8; ++o) acc[o] = eb[o];
  const float* srow = skip + (size_t)r*CH;
  for (int c = 0; c < CH; ++c) {
    float s = srow[c];
    #pragma unroll
    for (int o = 0; o < 8; ++o) acc[o] += s * ew[c*8 + o];
  }
  #pragma unroll
  for (int o = 0; o < 8; ++o) out[(size_t)r*8 + o] = acc[o];
}

// ---------------- host ----------------
extern "C" void kernel_launch(void* const* d_in, const int* in_sizes, int n_in,
                              void* d_out, int out_size, void* d_ws, size_t ws_size,
                              hipStream_t stream) {
  const float* audio     = (const float*)d_in[0];
  const float* spect     = (const float*)d_in[1];
  const float* start_w   = (const float*)d_in[2];
  const float* start_b   = (const float*)d_in[3];
  const float* in_w      = (const float*)d_in[4];
  const float* in_b      = (const float*)d_in[5];
  const float* cond_w    = (const float*)d_in[6];
  const float* cond_b    = (const float*)d_in[7];
  const float* rs_w      = (const float*)d_in[8];
  const float* rs_b      = (const float*)d_in[9];
  const float* rs_last_w = (const float*)d_in[10];
  const float* rs_last_b = (const float*)d_in[11];
  const float* end_w     = (const float*)d_in[12];
  const float* end_b     = (const float*)d_in[13];
  float* out = (float*)d_out;

  // workspace carve (aligned)
  size_t off = 0;
  char* base = (char*)d_ws;
  auto carve = [&](size_t bytes) { void* p = base + off; off += (bytes + 255) & ~(size_t)255; return p; };
  half_t* spect_h = (half_t*)carve((size_t)RTOT * MEL * sizeof(half_t));
  half_t* x0      = (half_t*)carve((size_t)RTOT * CH  * sizeof(half_t));
  half_t* x1      = (half_t*)carve((size_t)RTOT * CH  * sizeof(half_t));
  float*  skip    = (float*) carve((size_t)RTOT * CH  * sizeof(float));
  half_t* WtIn    = (half_t*)carve((size_t)NLAYERS * NOUT * KA * sizeof(half_t));
  half_t* WtRs    = (half_t*)carve((size_t)NLAYERS * NOUT * CH * sizeof(half_t));
  (void)ws_size; (void)in_sizes; (void)n_in; (void)out_size;

  // prep
  {
    long long n = (long long)RTOT * MEL;
    wg_cvt_spect<<<(unsigned)((n + 255) / 256), 256, 0, stream>>>(spect, spect_h, n);
  }
  {
    long long n = (long long)NLAYERS * NOUT * KA;
    wg_prep_wtin<<<(unsigned)((n + 255) / 256), 256, 0, stream>>>(in_w, cond_w, WtIn);
  }
  {
    long long n = (long long)NLAYERS * NOUT * CH;
    wg_prep_wtrs<<<(unsigned)((n + 255) / 256), 256, 0, stream>>>(rs_w, rs_last_w, WtRs);
  }
  {
    long long n = (long long)RTOT * CH;
    wg_start<<<(unsigned)((n + 255) / 256), 256, 0, stream>>>(audio, start_w, start_b, x0);
  }

  // layers (ping-pong x to keep dilated reads race-free)
  half_t* xin = x0;
  half_t* xout = x1;
  for (int i = 0; i < NLAYERS; ++i) {
    int dil = 1 << i;
    int is_last  = (i == NLAYERS - 1) ? 1 : 0;
    int is_first = (i == 0) ? 1 : 0;
    const float* rsb = is_last ? rs_last_b : (rs_b + (size_t)i * NOUT);
    wg_layer<<<RTOT / 32, 256, 0, stream>>>(
        xin, xout, spect_h,
        WtIn + (size_t)i * NOUT * KA,
        WtRs + (size_t)i * NOUT * CH,
        in_b  + (size_t)i * NOUT,
        cond_b + (size_t)i * NOUT,
        rsb, skip, dil, is_first, is_last);
    half_t* tmp = xin; xin = xout; xout = tmp;
  }

  wg_end<<<(RTOT + 255) / 256, 256, 0, stream>>>(skip, end_w, end_b, out);
}
